// Block_3453153706842
// MI455X (gfx1250) — matrix operand order repaired and checked by reading, not compiled, not run on hardware
//
#include <hip/hip_runtime.h>

// ---------- CDNA5 WMMA types ----------
typedef __attribute__((ext_vector_type(16))) __bf16 v16bf;
typedef __attribute__((ext_vector_type(8)))  float  v8f;
typedef __attribute__((ext_vector_type(4)))  unsigned uint32x4;
typedef __attribute__((ext_vector_type(8)))  unsigned uint32x8;

union BF16x16 { uint4 u4[2]; v16bf v; };

__device__ inline v8f v8f_zero() {
  v8f z;
#pragma unroll
  for (int i = 0; i < 8; ++i) z[i] = 0.f;
  return z;
}

// A-fragment (16x32 bf16, MxK): lane l (M = l&15, g = l>>4) holds
//   e0..7  -> k = 8g + e        (16B chunk)
//   e8..15 -> k = 16 + 8g + (e-8)
__device__ inline v16bf frag_ld_a(const __bf16* rowp, int g) {
  BF16x16 t;
  t.u4[0] = *(const uint4*)(rowp + 8 * g);
  t.u4[1] = *(const uint4*)(rowp + 16 + 8 * g);
  return t.v;
}

// B-fragment (32x16 bf16, KxN) read from an [n][k] (transposed) layout:
// lane l (N = l&15, g = l>>4) holds e0..7 -> k = 8g + e, e8..15 -> k = 16 + 8g + (e-8).
__device__ inline v16bf frag_ld_b(const __bf16* rowp, int g) {
  BF16x16 t;
  t.u4[0] = *(const uint4*)(rowp + 8 * g);
  t.u4[1] = *(const uint4*)(rowp + 16 + 8 * g);
  return t.v;
}

// ---------- CDNA5 async global->LDS copy (ASYNCcnt-tracked) ----------
// dst_lds: wave-relative LDS byte address (low 32 bits of a generic LDS ptr,
// per ISA 10.2 the LDS aperture maps addr[31:0] to the LDS address).
__device__ inline void async_copy_b128(void* dst_lds, const void* src_global) {
  unsigned lds_addr = (unsigned)(uintptr_t)dst_lds;
  asm volatile("global_load_async_to_lds_b128 %0, %1, off"
               :: "v"(lds_addr), "v"(src_global)
               : "memory");
}
__device__ inline void wait_asynccnt0() {
  asm volatile("s_wait_asynccnt 0x0" ::: "memory");
}

// ---------- CDNA5 Tensor Data Mover: 2D tile global -> LDS ----------
// Builds a 2-group D# (groups 2/3 NULL => up-to-2D tensor) per ISA 8.3/8.4.
// Moves nrows x rowlen(bf16) with row stride stride_elems into LDS, letting
// the TDM insert 16B of LDS padding after every 64B row (=> LDS stride 40
// bf16, matching frag_ld_a's layout). Tracked on TENSORcnt.
__device__ inline void tdm_load_2d_bf16(unsigned lds_addr, const void* gaddr,
                                        unsigned rowlen, unsigned nrows,
                                        unsigned stride_elems,
                                        unsigned tensor_d0, unsigned tensor_d1) {
  unsigned long long ga = (unsigned long long)(uintptr_t)gaddr;
  uint32x4 g0;
  g0[0] = 1u;                                   // count=1, user descriptor
  g0[1] = lds_addr;                             // LDS byte address
  g0[2] = (unsigned)(ga & 0xFFFFFFFFu);         // global_addr[31:0]
  g0[3] = (unsigned)((ga >> 32) & 0x01FFFFFFu)  // global_addr[56:32]
          | (2u << 30);                         // type=2 ("image")
  uint32x8 g1;
  g1[0] = (1u << 16)    // data_size: 2 bytes
        | (1u << 20)    // pad_enable
        | (3u << 22)    // pad_interval: 16 DWORDs (64B) between pads
        | (3u << 25);   // pad_amount: 4 DWORDs (16B)
  g1[1] = (tensor_d0 & 0xFFFFu) << 16;                      // tensor_dim0 lo
  g1[2] = (tensor_d0 >> 16) | ((tensor_d1 & 0xFFFFu) << 16);// dim0 hi | dim1 lo
  g1[3] = (tensor_d1 >> 16) | (rowlen << 16);               // dim1 hi | tile_dim0
  g1[4] = nrows;                                            // tile_dim1 (dim2=0)
  g1[5] = stride_elems;                                     // tensor_dim0_stride lo
  g1[6] = 0;                                                // stride0 hi | stride1 lo
  g1[7] = 0;
  asm volatile("tensor_load_to_lds %0, %1" :: "s"(g0), "s"(g1) : "memory");
}

// ---------- f32 -> bf16 conversion (vectorized) ----------
__global__ void __launch_bounds__(256) f32_to_bf16_kernel(
    const float* __restrict__ src, __bf16* __restrict__ dst, long long n4) {
  long long i = (long long)blockIdx.x * blockDim.x + threadIdx.x;
  long long stride = (long long)gridDim.x * blockDim.x;
  for (; i < n4; i += stride) {
    float4 f = ((const float4*)src)[i];
    union { __bf16 b[4]; uint2 u; } o;
    o.b[0] = (__bf16)f.x;
    o.b[1] = (__bf16)f.y;
    o.b[2] = (__bf16)f.z;
    o.b[3] = (__bf16)f.w;
    ((uint2*)dst)[i] = o.u;
  }
}

// ---------- RMSNorm (row = 2048 f32) -> bf16 ----------
__global__ void __launch_bounds__(256) rmsnorm_bf16_kernel(
    const float* __restrict__ x, const float* __restrict__ g,
    __bf16* __restrict__ y, int D) {
  int row = blockIdx.x;
  const float* xr = x + (size_t)row * D;
  float ss = 0.f;
  for (int i = threadIdx.x; i < D; i += blockDim.x) {
    float v = xr[i];
    ss += v * v;
  }
#pragma unroll
  for (int off = 16; off > 0; off >>= 1) ss += __shfl_xor(ss, off, 32);
  __shared__ float wsum[8];
  int wid = threadIdx.x >> 5, lane = threadIdx.x & 31;
  if (lane == 0) wsum[wid] = ss;
  __syncthreads();
  float tot = 0.f;
#pragma unroll
  for (int i = 0; i < 8; ++i) tot += wsum[i];
  float r = rsqrtf(tot / (float)D + 1e-6f);
  __bf16* yr = y + (size_t)row * D;
  for (int i = threadIdx.x; i < D; i += blockDim.x)
    yr[i] = (__bf16)(xr[i] * r * g[i]);
}

// ---------- bf16 WMMA GEMM: C[M,N] = A[M,K] * B[K,N] (+ fused epilogue) ----------
// A tile: TDM tensor_load_to_lds (one issue per workgroup, TENSORcnt).
// B tile: async global->LDS raw (ASYNCcnt), then LDS->LDS transpose to [n][k].
// EPI 0: outf = acc
// EPI 1: outb = bf16(acc)
// EPI 2: outf = aux + acc            (residual add)
// EPI 3: outb = bf16(silu(aux)*acc)  (FFN gate)
template <int EPI>
__global__ void __launch_bounds__(256) gemm_bf16_kernel(
    const __bf16* __restrict__ A, int lda,
    const __bf16* __restrict__ B, int ldb,
    const float* __restrict__ aux,
    float* __restrict__ outf, __bf16* __restrict__ outb, int ldc, int K) {
  __shared__ __attribute__((aligned(16))) __bf16 As[128 * 40];    // [m][k] (TDM-padded)
  __shared__ __attribute__((aligned(16))) __bf16 Bs[128 * 40];    // [n][k]
  __shared__ __attribute__((aligned(16))) __bf16 Braw[32 * 128];  // raw B tile

  int n0 = blockIdx.x * 128, m0 = blockIdx.y * 128;
  int tid = threadIdx.x;
  int wave = tid >> 5, lane = tid & 31, g = lane >> 4, l16 = lane & 15;
  int wm = wave >> 2, wn = wave & 3;  // 2x4 wave grid: 64x32 per wave

  v8f acc[4][2];
#pragma unroll
  for (int mt = 0; mt < 4; ++mt)
#pragma unroll
    for (int nt = 0; nt < 2; ++nt) acc[mt][nt] = v8f_zero();

  for (int k0 = 0; k0 < K; k0 += 32) {
    __syncthreads();  // LDS from previous iteration fully consumed
    // A tile 128x32 -> LDS via Tensor Data Mover (one descriptor, wave 0)
    if (wave == 0) {
      tdm_load_2d_bf16((unsigned)(uintptr_t)(void*)As,
                       A + (size_t)m0 * lda + k0,
                       /*rowlen=*/32u, /*nrows=*/128u,
                       /*stride=*/(unsigned)lda,
                       /*tensor_d0=*/(unsigned)K, /*tensor_d1=*/4096u);
    }
    // B tile 32x128 raw -> LDS via async DMA (no VGPR round trip)
#pragma unroll
    for (int i = 0; i < 2; ++i) {
      int c = tid + 256 * i;  // 0..511 chunks of 8 bf16
      int kk = c >> 4, ncol = (c & 15) * 8;
      async_copy_b128(Braw + kk * 128 + ncol,
                      B + (size_t)(k0 + kk) * ldb + n0 + ncol);
    }
    if (k0 + 32 < K) {  // global_prefetch_b8 for next K tile
      __builtin_prefetch(A + (size_t)(m0 + (tid >> 2)) * lda + k0 + 32, 0, 1);
      __builtin_prefetch(B + (size_t)(k0 + 32 + (tid >> 4)) * ldb + n0, 0, 1);
    }
    wait_asynccnt0();                          // Braw landed (per-wave issues)
    if (wave == 0) __builtin_amdgcn_s_wait_tensorcnt(0);  // As landed
    __syncthreads();

    // transpose Braw [k][n] -> Bs [n][k] (LDS->LDS, fragment-friendly layout)
#pragma unroll
    for (int i = 0; i < 2; ++i) {
      int c = tid + 256 * i;
      int kk = c >> 4, ncol = (c & 15) * 8;
      union { uint4 u; __bf16 b[8]; } v;
      v.u = *(const uint4*)(Braw + kk * 128 + ncol);
#pragma unroll
      for (int j = 0; j < 8; ++j) Bs[(ncol + j) * 40 + kk] = v.b[j];
    }
    __syncthreads();

    v16bf af[4];
#pragma unroll
    for (int mt = 0; mt < 4; ++mt)
      af[mt] = frag_ld_a(As + (size_t)(wm * 64 + mt * 16 + l16) * 40, g);
#pragma unroll
    for (int nt = 0; nt < 2; ++nt) {
      v16bf bfr = frag_ld_b(Bs + (size_t)(wn * 32 + nt * 16 + l16) * 40, g);
#pragma unroll
      for (int mt = 0; mt < 4; ++mt)
        acc[mt][nt] = __builtin_amdgcn_wmma_f32_16x16x32_bf16(
            false, af[mt], false, bfr, (short)0, acc[mt][nt], false, false);
    }
  }

  // epilogue: C layout -> lane holds col = l16, rows r + 8g per VGPR r
#pragma unroll
  for (int mt = 0; mt < 4; ++mt)
#pragma unroll
    for (int nt = 0; nt < 2; ++nt)
#pragma unroll
      for (int r = 0; r < 8; ++r) {
        int row = m0 + wm * 64 + mt * 16 + r + 8 * g;
        int col = n0 + wn * 32 + nt * 16 + l16;
        size_t idx = (size_t)row * ldc + col;
        float v = acc[mt][nt][r];
        if constexpr (EPI == 0) {
          outf[idx] = v;
        } else if constexpr (EPI == 1) {
          outb[idx] = (__bf16)v;
        } else if constexpr (EPI == 2) {
          outf[idx] = aux[idx] + v;
        } else {
          float t = aux[idx];
          float s = t / (1.f + __expf(-t));
          outb[idx] = (__bf16)(s * v);
        }
      }
}

// ---------- Flash attention (causal), bf16 WMMA ----------
// qkv: [B,S,3*H*Hd] bf16 rows (stride 6144); out: [B,S,2048] bf16
#define ATT_LDV 72
__global__ void __launch_bounds__(128) attention_kernel(
    const __bf16* __restrict__ qkv, __bf16* __restrict__ out) {
  __shared__ __attribute__((aligned(16))) __bf16 Vt[128 * ATT_LDV];      // [d][key]
  __shared__ __attribute__((aligned(16))) __bf16 Pscr[4 * 16 * ATT_LDV]; // per-wave P

  const int S = 2048, Hd = 128, ROWSTR = 6144;
  int qb = blockIdx.x, h = blockIdx.y, bb = blockIdx.z;
  int tid = threadIdx.x;
  int w = tid >> 5, lane = tid & 31, g = lane >> 4, l16 = lane & 15;
  int q0 = qb * 64;

  // Q fragments straight from global (rows are A-layout-friendly)
  const __bf16* qbase =
      qkv + ((size_t)(bb * S + q0 + w * 16 + l16)) * ROWSTR + h * Hd;
  v16bf qf[4];
#pragma unroll
  for (int ks = 0; ks < 4; ++ks) qf[ks] = frag_ld_a(qbase + 32 * ks, g);

  v8f o[8];
#pragma unroll
  for (int i = 0; i < 8; ++i) o[i] = v8f_zero();
  float mrow[8], lrow[8];
#pragma unroll
  for (int r = 0; r < 8; ++r) { mrow[r] = -1e30f; lrow[r] = 0.f; }
  const float scale = 0.08838834764831845f;  // 1/sqrt(128)

  int nkb = qb + 1;  // causal: key blocks 0..qb
  for (int kb = 0; kb < nkb; ++kb) {
    int k0 = kb * 64;
    __syncthreads();  // Vt / Pscr safe to overwrite
    // stage V^T: [d][key]
    {
      const __bf16* vbase =
          qkv + ((size_t)(bb * S + k0)) * ROWSTR + 4096 + h * Hd;
#pragma unroll
      for (int i = 0; i < 8; ++i) {
        int c = tid + 128 * i;  // 0..1023
        int key = c >> 4, dcol = (c & 15) * 8;
        union { uint4 u; __bf16 b[8]; } vv;
        vv.u = *(const uint4*)(vbase + (size_t)key * ROWSTR + dcol);
#pragma unroll
        for (int j = 0; j < 8; ++j) Vt[(dcol + j) * ATT_LDV + key] = vv.b[j];
      }
    }
    // S = Q K^T  (K rows in global already match the [n][k] B layout)
    v8f s[4];
#pragma unroll
    for (int nt = 0; nt < 4; ++nt) s[nt] = v8f_zero();
#pragma unroll
    for (int ks = 0; ks < 4; ++ks) {
#pragma unroll
      for (int nt = 0; nt < 4; ++nt) {
        const __bf16* kbase =
            qkv + ((size_t)(bb * S + k0 + nt * 16 + l16)) * ROWSTR + 2048 +
            h * Hd;
        v16bf bk = frag_ld_b(kbase + 32 * ks, g);
        s[nt] = __builtin_amdgcn_wmma_f32_16x16x32_bf16(
            false, qf[ks], false, bk, (short)0, s[nt], false, false);
      }
    }
    __syncthreads();  // Vt ready

    // scale + causal mask
#pragma unroll
    for (int nt = 0; nt < 4; ++nt) {
      int key = k0 + nt * 16 + l16;
#pragma unroll
      for (int r = 0; r < 8; ++r) {
        int row = q0 + w * 16 + r + 8 * g;
        float v = s[nt][r] * scale;
        s[nt][r] = (key <= row) ? v : -1e30f;
      }
    }
    // online softmax (rows replicated over 16 N-lanes; butterfly over them)
    float alpha[8];
#pragma unroll
    for (int r = 0; r < 8; ++r) {
      float mx = s[0][r];
#pragma unroll
      for (int nt = 1; nt < 4; ++nt) mx = fmaxf(mx, s[nt][r]);
#pragma unroll
      for (int off = 1; off < 16; off <<= 1)
        mx = fmaxf(mx, __shfl_xor(mx, off, 32));
      float mnew = fmaxf(mrow[r], mx);
      float a = __expf(mrow[r] - mnew);
      mrow[r] = mnew;
      alpha[r] = a;
      float sum = 0.f;
#pragma unroll
      for (int nt = 0; nt < 4; ++nt) {
        float p = __expf(s[nt][r] - mnew);
        s[nt][r] = p;
        sum += p;
      }
#pragma unroll
      for (int off = 1; off < 16; off <<= 1) sum += __shfl_xor(sum, off, 32);
      lrow[r] = lrow[r] * a + sum;
    }
#pragma unroll
    for (int i = 0; i < 8; ++i)
#pragma unroll
      for (int r = 0; r < 8; ++r) o[i][r] *= alpha[r];

    // P (C layout) -> LDS -> A layout
    __bf16* Pw = Pscr + w * 16 * ATT_LDV;
#pragma unroll
    for (int nt = 0; nt < 4; ++nt)
#pragma unroll
      for (int r = 0; r < 8; ++r)
        Pw[(r + 8 * g) * ATT_LDV + nt * 16 + l16] = (__bf16)s[nt][r];
    __syncthreads();

    // O += P * V
#pragma unroll
    for (int ks = 0; ks < 2; ++ks) {
      v16bf pa = frag_ld_a(Pw + (size_t)l16 * ATT_LDV + 32 * ks, g);
#pragma unroll
      for (int nt = 0; nt < 8; ++nt) {
        v16bf bv =
            frag_ld_b(Vt + (size_t)(nt * 16 + l16) * ATT_LDV + 32 * ks, g);
        o[nt] = __builtin_amdgcn_wmma_f32_16x16x32_bf16(
            false, pa, false, bv, (short)0, o[nt], false, false);
      }
    }
  }

  // normalize + store bf16 [B,S,2048]
#pragma unroll
  for (int nt = 0; nt < 8; ++nt)
#pragma unroll
    for (int r = 0; r < 8; ++r) {
      int row = q0 + w * 16 + r + 8 * g;
      int col = h * Hd + nt * 16 + l16;
      out[((size_t)(bb * S) + row) * 2048 + col] = (__bf16)(o[nt][r] / lrow[r]);
    }
}

// ---------- launcher ----------
extern "C" void kernel_launch(void* const* d_in, const int* in_sizes, int n_in,
                              void* d_out, int out_size, void* d_ws,
                              size_t ws_size, hipStream_t stream) {
  const float* x = (const float*)d_in[0];
  const float* w_qkv = (const float*)d_in[1];
  const float* w_out = (const float*)d_in[2];
  const float* w1 = (const float*)d_in[3];
  const float* w2 = (const float*)d_in[4];
  const float* w3 = (const float*)d_in[5];
  const float* g1 = (const float*)d_in[6];
  const float* g2 = (const float*)d_in[7];
  float* out = (float*)d_out;

  const int D = 2048, DFF = 8192, R = 4096;  // R = B*S rows

  char* p = (char*)d_ws;
  auto carve = [&](size_t bytes) -> char* {
    char* q = p;
    p += (bytes + 255) & ~(size_t)255;
    return q;
  };
  __bf16* wqkv_bf = (__bf16*)carve((size_t)D * 3 * D * 2);
  __bf16* wout_bf = (__bf16*)carve((size_t)D * D * 2);
  __bf16* w1_bf = (__bf16*)carve((size_t)D * DFF * 2);
  __bf16* w2_bf = (__bf16*)carve((size_t)DFF * D * 2);
  __bf16* w3_bf = (__bf16*)carve((size_t)D * DFF * 2);
  __bf16* xn_bf = (__bf16*)carve((size_t)R * D * 2);
  __bf16* qkv_bf = (__bf16*)carve((size_t)R * 3 * D * 2);
  __bf16* attn_bf = (__bf16*)carve((size_t)R * D * 2);
  float* t1 = (float*)carve((size_t)R * DFF * 4);
  __bf16* h_bf = (__bf16*)carve((size_t)R * DFF * 2);

  // weights f32 -> bf16 (bandwidth-trivial at 23.3 TB/s)
  f32_to_bf16_kernel<<<2048, 256, 0, stream>>>(w_qkv, wqkv_bf, (long long)D * 3 * D / 4);
  f32_to_bf16_kernel<<<2048, 256, 0, stream>>>(w_out, wout_bf, (long long)D * D / 4);
  f32_to_bf16_kernel<<<2048, 256, 0, stream>>>(w1, w1_bf, (long long)D * DFF / 4);
  f32_to_bf16_kernel<<<2048, 256, 0, stream>>>(w2, w2_bf, (long long)DFF * D / 4);
  f32_to_bf16_kernel<<<2048, 256, 0, stream>>>(w3, w3_bf, (long long)D * DFF / 4);

  // x1 = rmsnorm(x, g1)
  rmsnorm_bf16_kernel<<<R, 256, 0, stream>>>(x, g1, xn_bf, D);
  // qkv = x1 @ w_qkv        [4096 x 6144]
  gemm_bf16_kernel<1><<<dim3(48, 32), 256, 0, stream>>>(
      xn_bf, D, wqkv_bf, 3 * D, nullptr, nullptr, qkv_bf, 3 * D, D);
  // attn = causal flash attention(qkv)
  attention_kernel<<<dim3(32, 16, 2), 128, 0, stream>>>(qkv_bf, attn_bf);
  // out = x + attn @ w_out
  gemm_bf16_kernel<2><<<dim3(16, 32), 256, 0, stream>>>(
      attn_bf, D, wout_bf, D, x, out, nullptr, D, D);
  // x2 = rmsnorm(out, g2)
  rmsnorm_bf16_kernel<<<R, 256, 0, stream>>>(out, g2, xn_bf, D);
  // t1 = x2 @ w1            [4096 x 8192] f32
  gemm_bf16_kernel<0><<<dim3(64, 32), 256, 0, stream>>>(
      xn_bf, D, w1_bf, DFF, nullptr, t1, nullptr, DFF, D);
  // h = silu(t1) * (x2 @ w3)
  gemm_bf16_kernel<3><<<dim3(64, 32), 256, 0, stream>>>(
      xn_bf, D, w3_bf, DFF, t1, nullptr, h_bf, DFF, D);
  // out = out + h @ w2
  gemm_bf16_kernel<2><<<dim3(16, 32), 256, 0, stream>>>(
      h_bf, DFF, w2_bf, D, out, out, nullptr, D, DFF);
}
